// EEGFeatureExtractor_53687091200660
// MI455X (gfx1250) — matrix operand order repaired and checked by reading, not compiled, not run on hardware
//
#include <hip/hip_runtime.h>
#include <cstdint>

// MI455X / gfx1250: wave32, WMMA 16x16x32 bf16 (f32 accum), 320KB WGP LDS, TDM weight staging.

typedef __attribute__((ext_vector_type(16))) __bf16 v16bf;
typedef __attribute__((ext_vector_type(8)))  float  v8f;
typedef unsigned int v4u  __attribute__((ext_vector_type(4)));
typedef int          v8i_t __attribute__((ext_vector_type(8)));
typedef int          v4i_t __attribute__((ext_vector_type(4)));

#define R_CNT 14
#define BT 64
#define PITCH 272          // 136 bf16 per LDS row (128 data + 8 pad): 16B aligned, 4-bank row skew
#define OFF_WIH 0
#define OFF_WHH 139264     // 512*272
#define OFF_H   278528
#define OFF_X   295936
#define SMEM_BYTES 313344  // 2*512*272 + 2*64*272  (< 320KB)

// Tensor Data Mover path (6-arg builtin on this toolchain); falls back to copy loop if absent.
#if defined(__has_builtin)
#if __has_builtin(__builtin_amdgcn_tensor_load_to_lds) && __has_builtin(__builtin_amdgcn_s_wait_tensorcnt)
#define HAVE_TDM 1
#endif
#endif
#ifndef HAVE_TDM
#define HAVE_TDM 0
#endif

__device__ __constant__ int c_REGIDX[R_CNT][4] = {
  {0,1,16,17},{2,18,19,19},{3,4,4,4},{20,20,20,20},{7,8,8,8},{21,25,26,26},{5,22,22,22},
  {6,23,24,24},{9,27,27,27},{11,11,11,11},{29,29,29,29},{10,15,28,28},{12,30,30,30},{13,14,31,31}};
__device__ __constant__ int c_LENS[R_CNT] = {4,3,2,1,2,3,2,3,2,1,1,3,2,3};

union FragCast { struct { uint4 lo, hi; } q; v16bf v; };

__device__ __forceinline__ unsigned short f2bf(float f){   // RNE float->bf16
  unsigned int u = __float_as_uint(f);
  unsigned int rnd = 0x7FFFu + ((u >> 16) & 1u);
  return (unsigned short)((u + rnd) >> 16);
}
__device__ __forceinline__ float sigf(float v){ return 1.0f/(1.0f+__expf(-v)); }
__device__ __forceinline__ float tanhfast(float v){ return 2.0f/(1.0f+__expf(-2.0f*v)) - 1.0f; }

// A fragment (16x32 bf16, M=batch): lanes 0-15 rows m, K {k0..k0+7, k0+16..k0+23};
// lanes 16-31 same rows, K shifted by 8 (ISA 7.12.2 16-bit A table).
__device__ __forceinline__ v16bf load_fragA(const char* smem, int off, int mb, int k0, int lane){
  const int g = lane >> 4, m = lane & 15;
  const char* p = smem + off + (size_t)(mb*16 + m)*PITCH + (k0 + g*8)*2;
  FragCast f;
  f.q.lo = *(const uint4*)p;
  f.q.hi = *(const uint4*)(p + 32);
  return f.v;
}
// B fragment (32x16 bf16, K x N): lanes 0-15 col n K {k0..k0+7, k0+16..k0+23}; lanes 16-31 K shifted by 8.
__device__ __forceinline__ v16bf load_fragB(const char* smem, int off, int n0, int k0, int lane){
  const int g = lane >> 4, n = lane & 15;
  const char* p = smem + off + (size_t)(n0 + n)*PITCH + (k0 + g*8)*2;
  FragCast f;
  f.q.lo = *(const uint4*)p;
  f.q.hi = *(const uint4*)(p + 32);
  return f.v;
}

__device__ __forceinline__ void stage_x(char* smem, const float* __restrict__ x, int b0, int ts, int tid){
  for (int i = tid; i < BT*32; i += 256){
    int b = i >> 5, e4 = i & 31;
    const float4 f = *(const float4*)(x + ((size_t)(b0 + b)*32 + ts)*128 + e4*4);
    union { unsigned short h[4]; uint2 u; } pk;
    pk.h[0]=f2bf(f.x); pk.h[1]=f2bf(f.y); pk.h[2]=f2bf(f.z); pk.h[3]=f2bf(f.w);
    *(uint2*)(smem + OFF_X + (size_t)b*PITCH + e4*8) = pk.u;
  }
}

#if HAVE_TDM
// One TDM descriptor moves a whole 512x128 bf16 weight tile Global->LDS, with the
// hardware inserting 4 DWORDs of pad after every 64 DWORDs (PITCH=272 layout).
// D# per ISA ch.8: 2D tensor of 4-byte elements, dims 64 x 512, stride 64, pad enc (5,3).
__device__ __forceinline__ void tdm_load_tile(const void* gsrc, unsigned int lds_off){
  unsigned long long ga = (unsigned long long)(size_t)gsrc;
  v4u g0;
  g0[0] = 1u;                                                // count=1 (valid), no gather
  g0[1] = lds_off;                                           // lds_addr (bytes)
  g0[2] = (unsigned int)(ga & 0xFFFFFFFFu);                  // global_addr[31:0]
  g0[3] = (unsigned int)((ga >> 32) & 0x01FFFFFFu) | (2u<<30); // global_addr[56:32], type=2
  v8i_t g1;
  g1[0] = (int)((2u<<16) | (1u<<20) | (5u<<22) | (3u<<25));  // data_size=4B, pad_en, int=64dw, amt=4dw
  g1[1] = (int)(64u << 16);                                  // tensor_dim0[15:0]=64
  g1[2] = (int)(512u << 16);                                 // tensor_dim1[15:0]=512
  g1[3] = (int)(64u << 16);                                  // tile_dim0=64
  g1[4] = 512;                                               // tile_dim1=512 (tile_dim2=0)
  g1[5] = 64;                                                // tensor_dim0_stride=64
  g1[6] = 0;
  g1[7] = 0;
  v4i_t gz4; gz4[0]=0; gz4[1]=0; gz4[2]=0; gz4[3]=0;         // groups 2/3 unused (2D)
  v8i_t gz8;
  #pragma unroll
  for (int i = 0; i < 8; ++i) gz8[i] = 0;
  __builtin_amdgcn_tensor_load_to_lds(g0, g1, gz4, gz4, gz8, 0);
}
#endif

__device__ __forceinline__ void stage_weight_fallback(char* smem, unsigned int lds_off,
                                                      const unsigned short* gsrc, int tid){
  const uint4* s = (const uint4*)gsrc;
  for (int i = tid; i < 8192; i += 256){
    int row = i >> 4, p4 = i & 15;
    *(uint4*)(smem + lds_off + (size_t)row*PITCH + p4*16) = s[i];
  }
}

// ---------------- prep: permute+convert weights to bf16, fuse biases ----------------
// Wp[r][s][n][k], s: 0=W_ih fwd, 1=W_hh fwd, 2=W_ih bwd; column permutation
// n = (h/16)*64 + gate*16 + (h%16)  (orig row g0 = gate*128 + h) so each wave's 64-col
// chunk holds i,f,g,o tiles for the SAME 16 h-indices.
__global__ void prep_weights(const float* __restrict__ w_ih, const float* __restrict__ w_hh,
                             unsigned short* __restrict__ Wp){
  int idx = blockIdx.x*256 + threadIdx.x;              // < 14*3*65536
  int k    = idx & 127;
  int g0   = (idx >> 7) & 511;
  int slot = idx >> 16;                                // r*3 + s
  int r = slot/3, s = slot - r*3;
  float v;
  if      (s == 0) v = w_ih[((size_t)(r*2+0)*512 + g0)*128 + k];
  else if (s == 1) v = w_hh[((size_t)(r*2+0)*512 + g0)*128 + k];
  else             v = w_ih[((size_t)(r*2+1)*512 + g0)*128 + k];
  int gate = g0 >> 7, h = g0 & 127;
  int n = (h >> 4)*64 + gate*16 + (h & 15);
  Wp[((size_t)slot*512 + n)*128 + k] = f2bf(v);
}

__global__ void prep_bias(const float* __restrict__ b_ih, const float* __restrict__ b_hh,
                          float* __restrict__ biasP){
  int idx = blockIdx.x*256 + threadIdx.x;              // < 14*2*512 = 14336
  int g0 = idx & 511;
  int d  = (idx >> 9) & 1;
  int r  = idx >> 10;
  float v = b_ih[((size_t)(r*2+d) << 9) + g0] + b_hh[((size_t)(r*2+d) << 9) + g0];
  int gate = g0 >> 7, h = g0 & 127;
  int n = (h >> 4)*64 + gate*16 + (h & 15);
  biasP[((size_t)(r*2+d) << 9) + n] = v;
}

// ---------------- fused LSTM main kernel helpers ----------------
template<bool WITH_H>
__device__ __forceinline__ void gates_compute(const char* smem, int chunk, int lane,
                                              v8f (&acc)[4][4]){
  #pragma unroll
  for (int kk = 0; kk < 4; ++kk){
    const int k0 = kk*32;
    v16bf ax[4], ah[4];
    #pragma unroll
    for (int mb = 0; mb < 4; ++mb) ax[mb] = load_fragA(smem, OFF_X, mb, k0, lane);
    if (WITH_H){
      #pragma unroll
      for (int mb = 0; mb < 4; ++mb) ah[mb] = load_fragA(smem, OFF_H, mb, k0, lane);
    }
    #pragma unroll
    for (int q = 0; q < 4; ++q){
      v16bf bi = load_fragB(smem, OFF_WIH, chunk*64 + q*16, k0, lane);
      #pragma unroll
      for (int mb = 0; mb < 4; ++mb)
        acc[q][mb] = __builtin_amdgcn_wmma_f32_16x16x32_bf16(
            false, ax[mb], false, bi, (short)0, acc[q][mb], false, false);
      if (WITH_H){
        v16bf bh = load_fragB(smem, OFF_WHH, chunk*64 + q*16, k0, lane);
        #pragma unroll
        for (int mb = 0; mb < 4; ++mb)
          acc[q][mb] = __builtin_amdgcn_wmma_f32_16x16x32_bf16(
              false, ah[mb], false, bh, (short)0, acc[q][mb], false, false);
      }
    }
  }
}

__device__ __forceinline__ void lstm_update(const v8f (&acc)[4][4], const float (&bia)[4],
                                            float (&c_)[4][8], float (&hv)[4][8]){
  #pragma unroll
  for (int mb = 0; mb < 4; ++mb)
    #pragma unroll
    for (int j = 0; j < 8; ++j){
      float pi = acc[0][mb][j] + bia[0];
      float pf = acc[1][mb][j] + bia[1];
      float pg = acc[2][mb][j] + bia[2];
      float po = acc[3][mb][j] + bia[3];
      float cn = sigf(pf)*c_[mb][j] + sigf(pi)*tanhfast(pg);
      c_[mb][j] = cn;
      hv[mb][j] = sigf(po)*tanhfast(cn);
    }
}

__device__ __forceinline__ void store_h(char* smem, const float (&hv)[4][8],
                                        int chunk, int lane){
  const int g = lane >> 4, ln = lane & 15;
  const int col = chunk*16 + ln;
  #pragma unroll
  for (int mb = 0; mb < 4; ++mb)
    #pragma unroll
    for (int j = 0; j < 8; ++j){
      int m = mb*16 + j + g*8;
      *(unsigned short*)(smem + OFF_H + (size_t)m*PITCH + col*2) = f2bf(hv[mb][j]);
    }
}

// ---------------- main fused LSTM kernel ----------------
__global__ __launch_bounds__(256, 1)
void lstm_main(const float* __restrict__ x, const unsigned short* __restrict__ Wp,
               const float* __restrict__ biasP, float* __restrict__ out,
               float* __restrict__ part){
  __shared__ __align__(16) char smem[SMEM_BYTES];
  const int r  = blockIdx.y;
  const int b0 = blockIdx.x * BT;
  const int tid  = threadIdx.x;
  const int wave = tid >> 5, lane = tid & 31;
  const int g = lane >> 4, ln = lane & 15;
  const int chunk = wave;                 // wave owns h-indices [chunk*16, chunk*16+16)

  // ---- stage forward weights (TDM DMA with hardware row-padding, or copy fallback) ----
#if HAVE_TDM
  if (wave == 0){
    tdm_load_tile(Wp + (size_t)(r*3 + 0)*65536, OFF_WIH);
    tdm_load_tile(Wp + (size_t)(r*3 + 1)*65536, OFF_WHH);
  }
#else
  stage_weight_fallback(smem, OFF_WIH, Wp + (size_t)(r*3 + 0)*65536, tid);
  stage_weight_fallback(smem, OFF_WHH, Wp + (size_t)(r*3 + 1)*65536, tid);
#endif
  stage_x(smem, x, b0, c_REGIDX[r][0], tid);

  float bia[4];
  #pragma unroll
  for (int q = 0; q < 4; ++q) bia[q] = biasP[(size_t)(r*2 + 0)*512 + chunk*64 + q*16 + ln];

  float c_[4][8], hv[4][8];
  #pragma unroll
  for (int mb = 0; mb < 4; ++mb)
    #pragma unroll
    for (int j = 0; j < 8; ++j) c_[mb][j] = 0.0f;

  v8f vzero;
  #pragma unroll
  for (int j = 0; j < 8; ++j) vzero[j] = 0.0f;

  const int steps = c_LENS[r];

#if HAVE_TDM
  if (wave == 0) __builtin_amdgcn_s_wait_tensorcnt(0);
#endif

  // ---- t = 0 (h = 0: x-GEMM only) ----
  {
    __syncthreads();
    v8f acc[4][4];
    #pragma unroll
    for (int q = 0; q < 4; ++q)
      #pragma unroll
      for (int mb = 0; mb < 4; ++mb) acc[q][mb] = vzero;
    gates_compute<false>(smem, chunk, lane, acc);
    __syncthreads();
    if (steps > 1) stage_x(smem, x, b0, c_REGIDX[r][1], tid);
    lstm_update(acc, bia, c_, hv);
    if (steps > 1) store_h(smem, hv, chunk, lane);
  }
  // ---- t = 1..steps-1 (x-GEMM + h-GEMM) ----
  for (int t = 1; t < steps; ++t){
    __syncthreads();
    v8f acc[4][4];
    #pragma unroll
    for (int q = 0; q < 4; ++q)
      #pragma unroll
      for (int mb = 0; mb < 4; ++mb) acc[q][mb] = vzero;
    gates_compute<true>(smem, chunk, lane, acc);
    __syncthreads();
    if (t + 1 < steps) stage_x(smem, x, b0, c_REGIDX[r][t+1], tid);
    lstm_update(acc, bia, c_, hv);
    if (t + 1 < steps) store_h(smem, hv, chunk, lane);
  }

  // ---- backward: h0=c0=0 => gates = x_last @ W_ih_bwd + b; XBUF already holds x_last ----
#if HAVE_TDM
  if (wave == 0){
    tdm_load_tile(Wp + (size_t)(r*3 + 2)*65536, OFF_WIH);  // prior WIH reads done at last barrier
    __builtin_amdgcn_s_wait_tensorcnt(0);
  }
#else
  stage_weight_fallback(smem, OFF_WIH, Wp + (size_t)(r*3 + 2)*65536, tid);
#endif
  __syncthreads();

  float bia2[4];
  #pragma unroll
  for (int q = 0; q < 4; ++q) bia2[q] = biasP[(size_t)(r*2 + 1)*512 + chunk*64 + q*16 + ln];

  v8f acc[4][4];
  #pragma unroll
  for (int q = 0; q < 4; ++q)
    #pragma unroll
    for (int mb = 0; mb < 4; ++mb) acc[q][mb] = vzero;
  gates_compute<false>(smem, chunk, lane, acc);
  __syncthreads();                         // XBUF reads done; reuse it for stats scratch

  float hb[4][8];
  #pragma unroll
  for (int mb = 0; mb < 4; ++mb)
    #pragma unroll
    for (int j = 0; j < 8; ++j){
      float pi = acc[0][mb][j] + bia2[0];
      float pg = acc[2][mb][j] + bia2[2];
      float po = acc[3][mb][j] + bia2[3];
      float cn = sigf(pi)*tanhfast(pg);    // f*c0 == 0
      hb[mb][j] = sigf(po)*tanhfast(cn);
    }

  // write raw outputs + deterministic per-block partial sums for LayerNorm
  float s1 = 0.0f, s2 = 0.0f;
  const int col = chunk*16 + ln;
  #pragma unroll
  for (int mb = 0; mb < 4; ++mb)
    #pragma unroll
    for (int j = 0; j < 8; ++j){
      int m = mb*16 + j + g*8;
      size_t obase = ((size_t)(b0 + m)*R_CNT + r)*256;
      float vf = hv[mb][j], vb = hb[mb][j];
      out[obase + col]       = vf;
      out[obase + 128 + col] = vb;
      s1 += vf + vb;
      s2 += vf*vf + vb*vb;
    }
  #pragma unroll
  for (int o = 16; o > 0; o >>= 1){
    s1 += __shfl_xor(s1, o, 32);
    s2 += __shfl_xor(s2, o, 32);
  }
  float* red = (float*)(smem + OFF_X);
  if (lane == 0){ red[wave] = s1; red[8 + wave] = s2; }
  __syncthreads();
  if (tid == 0){
    float S1 = 0.0f, S2 = 0.0f;
    for (int w = 0; w < 8; ++w){ S1 += red[w]; S2 += red[8 + w]; }
    part[((size_t)r*128 + blockIdx.x)*2 + 0] = S1;
    part[((size_t)r*128 + blockIdx.x)*2 + 1] = S2;
  }
}

// ---------------- stats reduction (deterministic) + in-place normalize ----------------
__global__ void reduce_stats(const float* __restrict__ part, float* __restrict__ stats){
  int r = blockIdx.x, tid = threadIdx.x;                 // 128 threads
  float s1 = part[((size_t)r*128 + tid)*2 + 0];
  float s2 = part[((size_t)r*128 + tid)*2 + 1];
  #pragma unroll
  for (int o = 16; o > 0; o >>= 1){
    s1 += __shfl_xor(s1, o, 32);
    s2 += __shfl_xor(s2, o, 32);
  }
  __shared__ float tmp[8];
  int wave = tid >> 5, lane = tid & 31;
  if (lane == 0){ tmp[wave] = s1; tmp[4 + wave] = s2; }
  __syncthreads();
  if (tid == 0){
    float S1 = 0.0f, S2 = 0.0f;
    for (int w = 0; w < 4; ++w){ S1 += tmp[w]; S2 += tmp[4 + w]; }
    stats[r*2 + 0] = S1; stats[r*2 + 1] = S2;
  }
}

__global__ void normalize_k(float* __restrict__ out, const float* __restrict__ stats,
                            const float* __restrict__ gamma, const float* __restrict__ beta){
  int b = blockIdx.x, r = blockIdx.y, q = threadIdx.x;   // 64 threads * float4
  const float N = 8192.0f*256.0f;
  float mean = stats[r*2 + 0] / N;
  float var  = stats[r*2 + 1] / N - mean*mean;
  float scl  = rsqrtf(var + 1e-5f) * gamma[r];
  float bet  = beta[r];
  float4* p = (float4*)(out + ((size_t)b*R_CNT + r)*256) + q;
  float4 v = *p;
  v.x = (v.x - mean)*scl + bet;
  v.y = (v.y - mean)*scl + bet;
  v.z = (v.z - mean)*scl + bet;
  v.w = (v.w - mean)*scl + bet;
  *p = v;
}

extern "C" void kernel_launch(void* const* d_in, const int* in_sizes, int n_in,
                              void* d_out, int out_size, void* d_ws, size_t ws_size,
                              hipStream_t stream){
  const float* x     = (const float*)d_in[0];
  const float* w_ih  = (const float*)d_in[1];
  const float* w_hh  = (const float*)d_in[2];
  const float* b_ih  = (const float*)d_in[3];
  const float* b_hh  = (const float*)d_in[4];
  const float* gamma = (const float*)d_in[5];
  const float* beta  = (const float*)d_in[6];
  char* ws = (char*)d_ws;
  float*          part  = (float*)(ws + 0);      // 14*128*2*4 = 14336 B
  float*          stats = (float*)(ws + 14336);  // 112 B
  float*          biasP = (float*)(ws + 14592);  // 14*2*512*4 = 57344 B
  unsigned short* Wp    = (unsigned short*)(ws + 72192); // 14*3*512*128*2 = 5,505,024 B

  prep_weights<<<10752, 256, 0, stream>>>(w_ih, w_hh, Wp);   // 14*3*65536/256
  prep_bias   <<<56,    256, 0, stream>>>(b_ih, b_hh, biasP);
  lstm_main   <<<dim3(128, R_CNT), 256, 0, stream>>>(x, Wp, biasP, (float*)d_out, part);
  reduce_stats<<<R_CNT, 128, 0, stream>>>(part, stats);
  normalize_k <<<dim3(8192, R_CNT), 64, 0, stream>>>((float*)d_out, stats, gamma, beta);
}
